// SelfAttention_42588895707383
// MI455X (gfx1250) — compile-verified, matrix operand order checked by reading, not run on hardware
//
#include <hip/hip_runtime.h>
#include <math.h>
#include <stdint.h>

typedef __bf16 bf16;
typedef __attribute__((ext_vector_type(8)))  bf16  v8bf;
typedef __attribute__((ext_vector_type(16))) bf16  v16bf;
typedef __attribute__((ext_vector_type(8)))  float v8f;

#define B_      4
#define N_      2048
#define DIM_    1024
#define H_      16
#define HD_     64
#define PREFIX_ 4

// ---------------------------------------------------------------------------
// CDNA5 helpers
// ---------------------------------------------------------------------------

__device__ __forceinline__ v8f zero8() {
  v8f z;
#pragma unroll
  for (int i = 0; i < 8; ++i) z[i] = 0.0f;
  return z;
}

// Async copy 16 bytes global -> LDS (GLOBAL_LOAD_ASYNC_TO_LDS_B128, GV mode).
// VDST VGPR carries the per-lane LDS byte offset; ASYNCcnt tracks completion.
__device__ __forceinline__ void async_copy16(uint32_t lds_off, const void* g) {
  asm volatile("global_load_async_to_lds_b128 %0, %1, off"
               :
               : "v"(lds_off), "v"((unsigned long long)(uintptr_t)g)
               : "memory");
}

__device__ __forceinline__ void wait_async0() {
  asm volatile("s_wait_asynccnt 0" ::: "memory");
}

// Load a 16x32 (A) or 32-K x 16-N (B) bf16 fragment from a row-major
// [rows x stride] buffer (global or LDS).  ISA 7.12.2: lane r in [0,15]:
// row r0+r, K = {k0..k0+7, k0+16..k0+23}; lanes 16..31 shift kb by 8.
__device__ __forceinline__ v16bf load_frag(const bf16* base, int stride,
                                           int r0, int k0, int lane) {
  const int r  = r0 + (lane & 15);
  const int kb = k0 + ((lane >> 4) << 3);
  const bf16* p = base + (size_t)r * stride + kb;
  v8bf lo = *(const v8bf*)(p);
  v8bf hi = *(const v8bf*)(p + 16);
  v16bf out;
#pragma unroll
  for (int i = 0; i < 8; ++i) { out[i] = lo[i]; out[8 + i] = hi[i]; }
  return out;
}

__device__ __forceinline__ v8f wmma_bf16(v16bf a, v16bf b, v8f c) {
  return __builtin_amdgcn_wmma_f32_16x16x32_bf16(
      false, a, false, b, (short)0, c, false, false);
}

// ---------------------------------------------------------------------------
// Kernel 0: fp32 -> bf16 conversion
// ---------------------------------------------------------------------------
__global__ void f32_to_bf16_kernel(const float* __restrict__ src,
                                   bf16* __restrict__ dst, int n) {
  int i = blockIdx.x * blockDim.x + threadIdx.x;
  if (i < n) dst[i] = (bf16)src[i];
}

// ---------------------------------------------------------------------------
// Kernels 1 & 4: WMMA GEMM  C[M,Nout] = A[M,K] * Bw[Nout,K]^T
// Block: 256 threads = 8 waves, tile 128(M) x 128(N); wave tile 32x64.
// A/B k-slices (128x32 bf16 = 8KB each) staged via async global->LDS copies,
// double buffered: one s_wait_asynccnt + one barrier per k-step.
// MODE 0: out = f32 C + bias                       (projection)
// MODE 1: scatter bf16 q (pre-scaled by 1/sqrt(hd)), k, V^T    (QKV)
// ---------------------------------------------------------------------------
template <int MODE>
__global__ __launch_bounds__(256) void gemm_wmma_kernel(
    const bf16* __restrict__ A, const bf16* __restrict__ Bw,
    const float* __restrict__ bias, float* __restrict__ outF,
    bf16* __restrict__ qo, bf16* __restrict__ ko, bf16* __restrict__ vto,
    int M, int Nout, int K) {
  __shared__ __align__(16) bf16 Abuf[2][128 * 32];
  __shared__ __align__(16) bf16 Bbuf[2][128 * 32];

  const int tid  = threadIdx.x;
  const int lane = tid & 31;
  const int wave = tid >> 5;
  const int half = lane >> 4;
  const int r    = lane & 15;
  const int mblk = blockIdx.y * 128;
  const int nblk = blockIdx.x * 128;
  const int wm0  = (wave >> 1) * 32;   // wave tile origin within block
  const int wn0  = (wave & 1) * 64;

  // Stage one 128x32 slice of A and of Bw into LDS buffer `buf`.
  // 512 chunks of 16B per matrix; each thread issues 2+2 async copies.
  auto stage = [&](int buf, int ks) {
    const uint32_t la = (uint32_t)(uintptr_t)&Abuf[buf][0];
    const uint32_t lb = (uint32_t)(uintptr_t)&Bbuf[buf][0];
#pragma unroll
    for (int cc = 0; cc < 2; ++cc) {
      const int c   = tid + cc * 256;      // 0..511
      const int row = c >> 2;              // 128 rows
      const int off = (c & 3) * 16;        // 4 x 16B per 64B row slice
      const char* ga = (const char*)(A  + (size_t)(mblk + row) * K + ks) + off;
      const char* gb = (const char*)(Bw + (size_t)(nblk + row) * K + ks) + off;
      async_copy16(la + (uint32_t)c * 16, ga);
      async_copy16(lb + (uint32_t)c * 16, gb);
    }
  };

  v8f acc[2][4];
#pragma unroll
  for (int i = 0; i < 2; ++i)
#pragma unroll
    for (int j = 0; j < 4; ++j) acc[i][j] = zero8();

  stage(0, 0);
  for (int ks = 0; ks < K; ks += 32) {
    const int cur = (ks >> 5) & 1;
    wait_async0();          // my async copies into buf[cur] done
    __syncthreads();        // everyone's copies done; buf[cur^1] free
    if (ks + 32 < K) stage(cur ^ 1, ks + 32);

    const bf16* At = &Abuf[cur][0];
    const bf16* Bt = &Bbuf[cur][0];
    v16bf a0 = load_frag(At, 32, wm0,      0, lane);
    v16bf a1 = load_frag(At, 32, wm0 + 16, 0, lane);
#pragma unroll
    for (int j = 0; j < 4; ++j) {
      v16bf b = load_frag(Bt, 32, wn0 + 16 * j, 0, lane);
      acc[0][j] = wmma_bf16(a0, b, acc[0][j]);
      acc[1][j] = wmma_bf16(a1, b, acc[1][j]);
    }
  }

  const int m0 = mblk + wm0;
  const int n0 = nblk + wn0;
#pragma unroll
  for (int i = 0; i < 2; ++i) {
#pragma unroll
    for (int j = 0; j < 4; ++j) {
#pragma unroll
      for (int vg = 0; vg < 8; ++vg) {
        const int row = m0 + 16 * i + vg + 8 * half;  // C layout: M = vg(+8)
        const int col = n0 + 16 * j + r;              //           N = lane
        float v = acc[i][j][vg];
        if (MODE == 0) {
          outF[(size_t)row * Nout + col] = v + bias[col];
        } else {
          const int which = col >> 10;       // 0=q 1=k 2=v
          const int cc    = col & 1023;
          const int h     = cc >> 6;
          const int d     = cc & 63;
          const int b     = row >> 11;       // N_ = 2048
          const int n     = row & 2047;
          const size_t bh = (size_t)(b * H_ + h);
          if (which == 0) {
            qo[(bh * N_ + n) * HD_ + d] = (bf16)(v * 0.125f);  // fold 1/sqrt(64)
          } else if (which == 1) {
            ko[(bh * N_ + n) * HD_ + d] = (bf16)v;
          } else {
            vto[(bh * HD_ + d) * N_ + n] = (bf16)v;            // V^T
          }
        }
      }
    }
  }
}

// ---------------------------------------------------------------------------
// Kernel 2: RoPE in-place on bf16 q,k ([B,H,N,HD]); each thread owns a
// (d, d+32) pair so the rotate-half exchange is race-free.
// ---------------------------------------------------------------------------
__global__ void rope_kernel(bf16* __restrict__ q, bf16* __restrict__ k,
                            const float* __restrict__ sinT,
                            const float* __restrict__ cosT) {
  int tid = blockIdx.x * blockDim.x + threadIdx.x;
  if (tid >= B_ * H_ * N_ * 32) return;
  const int d  = tid & 31;
  const int n  = (tid >> 5) & (N_ - 1);
  const int bh = tid >> 16;             // 32 * 2048 = 2^16
  if (n < PREFIX_) return;
  const int rp = n - PREFIX_;
  const float s1 = sinT[rp * HD_ + d],      c1 = cosT[rp * HD_ + d];
  const float s2 = sinT[rp * HD_ + d + 32], c2 = cosT[rp * HD_ + d + 32];
  const size_t base = ((size_t)bh * N_ + n) * HD_;
  {
    float t1 = (float)q[base + d], t2 = (float)q[base + d + 32];
    q[base + d]      = (bf16)(t1 * c1 - t2 * s1);
    q[base + d + 32] = (bf16)(t2 * c2 + t1 * s2);
  }
  {
    float t1 = (float)k[base + d], t2 = (float)k[base + d + 32];
    k[base + d]      = (bf16)(t1 * c1 - t2 * s1);
    k[base + d + 32] = (bf16)(t2 * c2 + t1 * s2);
  }
}

// ---------------------------------------------------------------------------
// Kernel 3: flash attention.  1 block = 128 q-rows of one (b,h); 8 waves,
// each wave privately owns a 16-row stripe (full softmax row in-wave).
// K / V^T tiles (64x64 bf16 = 8KB each) staged via async global->LDS, double
// buffered.  S = Q K^T via WMMA; online softmax with 16-lane shfl reductions;
// P transposed to A-layout through wave-private LDS (in-order per wave);
// O += P V via WMMA.  Q is pre-scaled by 1/sqrt(hd) at the QKV epilogue.
// ---------------------------------------------------------------------------
__global__ __launch_bounds__(256) void flash_attn_kernel(
    const bf16* __restrict__ Q, const bf16* __restrict__ Kc,
    const bf16* __restrict__ VT, bf16* __restrict__ O) {
  __shared__ __align__(16) bf16 Kbuf[2][64 * 64];
  __shared__ __align__(16) bf16 Vbuf[2][64 * 64];
  __shared__ __align__(16) bf16 Pst[8][16 * 64];

  const int tid  = threadIdx.x;
  const int lane = tid & 31;
  const int wave = tid >> 5;
  const int half = lane >> 4;
  const int r    = lane & 15;
  const int qb   = blockIdx.x & 15;     // N_/128 = 16 q blocks
  const int bh   = blockIdx.x >> 4;     // 0..63
  const bf16* Qb = Q  + (size_t)bh * N_ * HD_;
  const bf16* Kb = Kc + (size_t)bh * N_ * HD_;
  const bf16* Vb = VT + (size_t)bh * HD_ * N_;
  const int m0 = qb * 128 + wave * 16;

  // Stage key tile [kt, kt+64) of K ([64 x 64], contiguous 8KB) and of
  // V^T ([64 d x 64 keys], 64 segments of 128B) into LDS buffer `buf`.
  auto stage = [&](int buf, int kt) {
    const uint32_t lk = (uint32_t)(uintptr_t)&Kbuf[buf][0];
    const uint32_t lv = (uint32_t)(uintptr_t)&Vbuf[buf][0];
    const char* gk = (const char*)(Kb + (size_t)kt * HD_);
    async_copy16(lk + (uint32_t)tid * 16,        gk + tid * 16);
    async_copy16(lk + (uint32_t)tid * 16 + 4096, gk + tid * 16 + 4096);
#pragma unroll
    for (int cc = 0; cc < 2; ++cc) {
      const int c   = tid + cc * 256;   // 0..511
      const int d   = c >> 3;
      const int seg = c & 7;
      const char* gv = (const char*)(Vb + (size_t)d * N_ + kt + seg * 8);
      async_copy16(lv + (uint32_t)c * 16, gv);
    }
  };

  const v16bf aq0 = load_frag(Qb, HD_, m0, 0,  lane);
  const v16bf aq1 = load_frag(Qb, HD_, m0, 32, lane);

  v8f  o[4];
  float mi[8], li[8];
#pragma unroll
  for (int j = 0; j < 4; ++j) o[j] = zero8();
#pragma unroll
  for (int vg = 0; vg < 8; ++vg) { mi[vg] = -3.0e38f; li[vg] = 0.0f; }

  stage(0, 0);
  for (int kt = 0; kt < N_; kt += 64) {
    const int cur = (kt >> 6) & 1;
    wait_async0();
    __syncthreads();
    if (kt + 64 < N_) stage(cur ^ 1, kt + 64);
    const bf16* Kt = &Kbuf[cur][0];
    const bf16* Vt = &Vbuf[cur][0];

    // ---- S = Q K^T (16 x 64 stripe, 8 WMMAs) ----
    v8f s[4];
#pragma unroll
    for (int j = 0; j < 4; ++j) {
      s[j] = zero8();
      v16bf bk0 = load_frag(Kt, 64, 16 * j, 0,  lane);
      s[j] = wmma_bf16(aq0, bk0, s[j]);
      v16bf bk1 = load_frag(Kt, 64, 16 * j, 32, lane);
      s[j] = wmma_bf16(aq1, bk1, s[j]);
    }
    // ---- online softmax (rows live in 16-lane halves) ----
#pragma unroll
    for (int vg = 0; vg < 8; ++vg) {
      float mx = -3.0e38f;
#pragma unroll
      for (int j = 0; j < 4; ++j) mx = fmaxf(mx, s[j][vg]);
#pragma unroll
      for (int m = 8; m >= 1; m >>= 1) mx = fmaxf(mx, __shfl_xor(mx, m, 32));
      const float nm    = fmaxf(mi[vg], mx);
      const float alpha = __expf(mi[vg] - nm);
      float rs = 0.0f;
#pragma unroll
      for (int j = 0; j < 4; ++j) {
        float p = __expf(s[j][vg] - nm);
        s[j][vg] = p;
        rs += p;
      }
#pragma unroll
      for (int m = 8; m >= 1; m >>= 1) rs += __shfl_xor(rs, m, 32);
      li[vg] = li[vg] * alpha + rs;
      mi[vg] = nm;
#pragma unroll
      for (int j = 0; j < 4; ++j) o[j][vg] *= alpha;
    }
    // ---- transpose P (C layout -> A layout) via wave-private LDS ----
    bf16* pst = &Pst[wave][0];
#pragma unroll
    for (int j = 0; j < 4; ++j)
#pragma unroll
      for (int vg = 0; vg < 8; ++vg)
        pst[(vg + 8 * half) * 64 + 16 * j + r] = (bf16)s[j][vg];
    // LDS is in-order per wave: stores above precede the loads below.
    v16bf ap0 = load_frag(pst, 64, 0, 0,  lane);
    v16bf ap1 = load_frag(pst, 64, 0, 32, lane);
    // ---- O += P V ----
#pragma unroll
    for (int j = 0; j < 4; ++j) {
      v16bf bv0 = load_frag(Vt, 64, 16 * j, 0,  lane);
      o[j] = wmma_bf16(ap0, bv0, o[j]);
      v16bf bv1 = load_frag(Vt, 64, 16 * j, 32, lane);
      o[j] = wmma_bf16(ap1, bv1, o[j]);
    }
  }

  // ---- normalize and write [B, N, H*HD] bf16 ----
  const int b = bh >> 4;
  const int h = bh & 15;
#pragma unroll
  for (int j = 0; j < 4; ++j) {
#pragma unroll
    for (int vg = 0; vg < 8; ++vg) {
      const int n_tok = m0 + vg + 8 * half;
      const int d     = 16 * j + r;
      O[((size_t)(b * N_ + n_tok)) * DIM_ + h * HD_ + d] =
          (bf16)(o[j][vg] / li[vg]);
    }
  }
}

// ---------------------------------------------------------------------------
// Host launcher
// ---------------------------------------------------------------------------
extern "C" void kernel_launch(void* const* d_in, const int* in_sizes, int n_in,
                              void* d_out, int out_size, void* d_ws,
                              size_t ws_size, hipStream_t stream) {
  const float* x     = (const float*)d_in[0];
  const float* Wqkv  = (const float*)d_in[1];
  const float* Wproj = (const float*)d_in[2];
  const float* bproj = (const float*)d_in[3];
  const float* sinT  = (const float*)d_in[4];
  const float* cosT  = (const float*)d_in[5];
  float* out = (float*)d_out;

  char*  ws  = (char*)d_ws;
  size_t off = 0;
  auto take = [&](size_t bytes) -> char* {
    char* p = ws + off;
    off += (bytes + 255) & ~(size_t)255;
    return p;
  };
  bf16* x_bf     = (bf16*)take((size_t)B_ * N_ * DIM_ * 2);
  bf16* wqkv_bf  = (bf16*)take((size_t)3 * DIM_ * DIM_ * 2);
  bf16* wproj_bf = (bf16*)take((size_t)DIM_ * DIM_ * 2);
  bf16* q_bf     = (bf16*)take((size_t)B_ * H_ * N_ * HD_ * 2);
  bf16* k_bf     = (bf16*)take((size_t)B_ * H_ * N_ * HD_ * 2);
  bf16* vt_bf    = (bf16*)take((size_t)B_ * H_ * N_ * HD_ * 2);
  bf16* attn_bf  = (bf16*)take((size_t)B_ * N_ * DIM_ * 2);

  const int nx = B_ * N_ * DIM_;
  const int nq = 3 * DIM_ * DIM_;
  const int np = DIM_ * DIM_;
  f32_to_bf16_kernel<<<(nx + 255) / 256, 256, 0, stream>>>(x, x_bf, nx);
  f32_to_bf16_kernel<<<(nq + 255) / 256, 256, 0, stream>>>(Wqkv, wqkv_bf, nq);
  f32_to_bf16_kernel<<<(np + 255) / 256, 256, 0, stream>>>(Wproj, wproj_bf, np);

  // QKV GEMM: [8192,1024] x [3072,1024]^T, scatter epilogue (q scaled)
  gemm_wmma_kernel<1><<<dim3(3 * DIM_ / 128, B_ * N_ / 128), 256, 0, stream>>>(
      x_bf, wqkv_bf, nullptr, nullptr, q_bf, k_bf, vt_bf,
      B_ * N_, 3 * DIM_, DIM_);

  const int nr = B_ * H_ * N_ * 32;
  rope_kernel<<<(nr + 255) / 256, 256, 0, stream>>>(q_bf, k_bf, sinT, cosT);

  flash_attn_kernel<<<B_ * H_ * (N_ / 128), 256, 0, stream>>>(
      q_bf, k_bf, vt_bf, attn_bf);

  // Projection GEMM + bias -> fp32 output
  gemm_wmma_kernel<0><<<dim3(DIM_ / 128, B_ * N_ / 128), 256, 0, stream>>>(
      attn_bf, wproj_bf, bproj, out, nullptr, nullptr, nullptr,
      B_ * N_, DIM_, DIM_);
}
